// Seq2SeqLSTM_Attn_86543591014545
// MI455X (gfx1250) — matrix operand order repaired and checked by reading, not compiled, not run on hardware
//
#include <hip/hip_runtime.h>
#include <hip/hip_bf16.h>

// ---------------------------------------------------------------------------
// Seq2Seq LSTM + attention for MI455X (gfx1250, wave32, WMMA).
// Strategy:
//   * bf16 operands + f32 accumulation via v_wmma_f32_16x16x32_bf16.
//   * Input projections hoisted into fat GEMMs (M = T*B) outside the scan.
//   * Weights pre-transposed to (K x N) so B-fragments (lane=n, 16 k values
//     per lane at stride ldb; adjacent lanes read adjacent n).
//   * fc_W bf16 = 64MB -> resident in 192MB L2 across the 127 decoder steps.
// ---------------------------------------------------------------------------

typedef __attribute__((ext_vector_type(16))) __bf16 v16bf;
typedef __attribute__((ext_vector_type(8)))  float  v8f;

static constexpr int Vv  = 32000;
static constexpr int Ee  = 512;
static constexpr int Hh  = 1024;
static constexpr int Bb  = 32;
static constexpr int Ss  = 128;
static constexpr int Tt  = 128;
static constexpr int G4H = 4 * Hh;   // 4096 gate width

// ----------------------------- bf16 helpers --------------------------------
__device__ __forceinline__ unsigned short f2bf(float f) {
  union { float f; unsigned int u; } v; v.f = f;
  unsigned int r = v.u + 0x7FFFu + ((v.u >> 16) & 1u);   // round-to-nearest-even
  return (unsigned short)(r >> 16);
}
__device__ __forceinline__ float bf2f(unsigned short h) {
  union { unsigned int u; float f; } v; v.u = ((unsigned int)h) << 16;
  return v.f;
}
__device__ __forceinline__ float sigmoidf(float x) {
  return 1.0f / (1.0f + __expf(-x));
}

// ----------------------- WMMA fragment loaders -----------------------------
// A-matrix 16x32 bf16 (M x K): lane<16 -> row=lane, K = k0+0..7 and k0+16..23
//                              lane>=16 -> row=lane-16, K = k0+8..15, k0+24..31
__device__ __forceinline__ v16bf load_afrag(const unsigned short* __restrict__ base,
                                            int lda) {
  const int lane = threadIdx.x & 31;
  const unsigned short* p = base + (long)(lane & 15) * lda + ((lane >> 4) << 3);
  union { v16bf v; uint4 q[2]; } f;
  f.q[0] = *(const uint4*)(p);
  f.q[1] = *(const uint4*)(p + 16);
  return f.v;
}
// B-matrix 32x16 bf16 (K x N): lane<16 -> n=n0+lane, K = k0+0..7, k0+16..23;
// lane>=16 -> n=n0+lane-16, K = k0+8..15, k0+24..31.
// Wt is (K x Ntot) row-major, base = Wt + k0*ldb + n0.
__device__ __forceinline__ v16bf load_bfrag(const unsigned short* __restrict__ base,
                                            int ldb) {
  const int lane = threadIdx.x & 31;
  const unsigned short* p = base + (lane & 15) + (long)((lane >> 4) << 3) * ldb;
  union { v16bf v; unsigned short u[16]; } f;
#pragma unroll
  for (int s = 0; s < 16; ++s) f.u[s] = p[(long)(s < 8 ? s : s + 8) * ldb];
  return f.v;
}

// ------------------------------ GEMM kernel --------------------------------
// C(M,N) [f32, ldc] (+)= A(M,K)[bf16,row-major,lda] * Wt(K,N)[bf16,row-major,ldb]
//                        (+ bias[n] if bias != nullptr)
// Block: 256 threads = 8 waves, 2x4 wave grid, wave tile 16x64, block 32x256.
// Requires M%32==0, N%256==0, K%32==0 (all shapes here satisfy this).
__global__ __launch_bounds__(256)
void gemm_wmma_bf16(const unsigned short* __restrict__ A, int lda,
                    const unsigned short* __restrict__ Wt, int ldb,
                    float* __restrict__ C, long ldc,
                    const float* __restrict__ bias, int accumulate, int K) {
  const int wave = threadIdx.x >> 5;
  const int wm = wave >> 2;                 // 0..1
  const int wn = wave & 3;                  // 0..3
  const int m0 = blockIdx.y * 32 + wm * 16;
  const int n0 = blockIdx.x * 256 + wn * 64;

  v8f acc[4] = {};
  const unsigned short* ap = A + (long)m0 * lda;
  const unsigned short* bp = Wt + n0;

  for (int k = 0; k < K; k += 32) {
    const unsigned short* bk = bp + (long)k * ldb;
    if (k + 32 < K) __builtin_prefetch(bp + (long)(k + 32) * ldb, 0, 1);
    v16bf a = load_afrag(ap + k, lda);
#pragma unroll
    for (int j = 0; j < 4; ++j) {
      v16bf b = load_bfrag(bk + j * 16, ldb);
      acc[j] = __builtin_amdgcn_wmma_f32_16x16x32_bf16(
          false, a, false, b, (short)0, acc[j], false, false);
    }
  }

  // C/D layout: lane<16 -> n=n0+lane, rows m0+0..7 ; lane>=16 -> n=n0+lane-16,
  // rows m0+8..15 (VGPR i <-> row offset i).
  const int lane = threadIdx.x & 31;
  const int mrow = m0 + ((lane >> 4) << 3);
  const int ncol = n0 + (lane & 15);
#pragma unroll
  for (int j = 0; j < 4; ++j) {
    const long n = (long)ncol + j * 16;
    const float badd = bias ? bias[n] : 0.0f;
    float* cp = C + (long)mrow * ldc + n;
#pragma unroll
    for (int i = 0; i < 8; ++i) {
      float v = acc[j][i] + badd;
      if (accumulate) v += cp[(long)i * ldc];
      cp[(long)i * ldc] = v;
    }
  }
}

// ------------------------ weight convert + transpose -----------------------
// in: (N x K) f32 row-major  ->  out: (K x N) bf16 row-major
__global__ void convert_transpose(const float* __restrict__ in,
                                  unsigned short* __restrict__ out,
                                  int N, int K, long total) {
  long idx = (long)blockIdx.x * blockDim.x + threadIdx.x;
  if (idx >= total) return;
  int n = (int)(idx / K);
  int k = (int)(idx % K);
  out[(long)k * N + n] = f2bf(in[idx]);
}

// ------------------------------ embedding ----------------------------------
// out[t][b][e] = bf16(embed[tok[b*tok_ld + t]][e]),  t in [0, steps)
__global__ void embed_lookup(const int* __restrict__ tok, int tok_ld,
                             const float* __restrict__ embed,
                             unsigned short* __restrict__ out, long total) {
  long idx = (long)blockIdx.x * blockDim.x + threadIdx.x;
  if (idx >= total) return;
  int e = (int)(idx % Ee);
  long tb = idx / Ee;
  int b = (int)(tb % Bb);
  int t = (int)(tb / Bb);
  int token = tok[b * tok_ld + t];
  out[idx] = f2bf(embed[(long)token * Ee + e]);
}

// --------------------------- LSTM pointwise --------------------------------
// gates G(b, 4H) pre-activations (x@Wih.T + h@Whh.T already summed in-place).
// i,f,g,o split order matches jnp.split. c updated in place (f32); h written
// as bf16 (consumed only by bf16 matmuls / attention).
__global__ void lstm_pointwise(const float* __restrict__ G,
                               const float* __restrict__ bih,
                               const float* __restrict__ bhh,
                               float* __restrict__ c,
                               unsigned short* __restrict__ h_out) {
  int idx = blockIdx.x * blockDim.x + threadIdx.x;       // b*H + j
  if (idx >= Bb * Hh) return;
  int b = idx / Hh, j = idx % Hh;
  const float* g = G + (long)b * G4H;
  float gi = g[j]          + bih[j]          + bhh[j];
  float gf = g[Hh + j]     + bih[Hh + j]     + bhh[Hh + j];
  float gg = g[2 * Hh + j] + bih[2 * Hh + j] + bhh[2 * Hh + j];
  float go = g[3 * Hh + j] + bih[3 * Hh + j] + bhh[3 * Hh + j];
  float cn = sigmoidf(gf) * c[idx] + sigmoidf(gi) * tanhf(gg);
  c[idx] = cn;
  h_out[idx] = f2bf(sigmoidf(go) * tanhf(cn));
}

// ----------------------------- attention -----------------------------------
// enc is time-major: enc[(s*B + b)*H + h]. One wave per (b,s) dot product.
__global__ void attn_scores(const unsigned short* __restrict__ enc,
                            const unsigned short* __restrict__ h,
                            float* __restrict__ scores) {
  int w = (blockIdx.x * blockDim.x + threadIdx.x) >> 5;
  int lane = threadIdx.x & 31;
  if (w >= Bb * Ss) return;
  int b = w / Ss, s = w % Ss;
  const unsigned short* e  = enc + ((long)s * Bb + b) * Hh;
  const unsigned short* hp = h + (long)b * Hh;
  float sum = 0.0f;
  for (int j = lane; j < Hh; j += 32) sum += bf2f(e[j]) * bf2f(hp[j]);
  for (int o = 16; o > 0; o >>= 1) sum += __shfl_down(sum, o, 32);
  if (lane == 0) scores[b * Ss + s] = sum;
}

__global__ void softmax128(const float* __restrict__ scores,
                           float* __restrict__ w) {
  __shared__ float sh[Ss];
  int b = blockIdx.x, s = threadIdx.x;
  float v = scores[b * Ss + s];
  sh[s] = v; __syncthreads();
  for (int o = 64; o > 0; o >>= 1) {
    if (s < o) sh[s] = fmaxf(sh[s], sh[s + o]);
    __syncthreads();
  }
  float mx = sh[0]; __syncthreads();
  float e = __expf(v - mx);
  sh[s] = e; __syncthreads();
  for (int o = 64; o > 0; o >>= 1) {
    if (s < o) sh[s] += sh[s + o];
    __syncthreads();
  }
  w[b * Ss + s] = e / sh[0];
}

// ctx[b][j] = sum_s w[b][s] * enc[s][b][j]   (f32 accumulate, bf16 out)
__global__ void attn_ctx(const float* __restrict__ w,
                         const unsigned short* __restrict__ enc,
                         unsigned short* __restrict__ ctx) {
  int idx = blockIdx.x * blockDim.x + threadIdx.x;       // b*H + j
  if (idx >= Bb * Hh) return;
  int b = idx / Hh, j = idx % Hh;
  const unsigned short* e = enc + (long)b * Hh + j;
  const float* wp = w + b * Ss;
  float sum = 0.0f;
  for (int s = 0; s < Ss; ++s) sum += wp[s] * bf2f(e[(long)s * Bb * Hh]);
  ctx[idx] = f2bf(sum);
}

// ------------------------------ host driver --------------------------------
extern "C" void kernel_launch(void* const* d_in, const int* in_sizes, int n_in,
                              void* d_out, int out_size, void* d_ws, size_t ws_size,
                              hipStream_t stream) {
  (void)in_sizes; (void)n_in; (void)out_size; (void)ws_size;

  const int*   src   = (const int*)d_in[0];
  const int*   tgt   = (const int*)d_in[1];
  const float* embed = (const float*)d_in[2];
  const float* Wih0  = (const float*)d_in[3];
  const float* Whh0  = (const float*)d_in[4];
  const float* bih0  = (const float*)d_in[5];
  const float* bhh0  = (const float*)d_in[6];
  const float* Wih1  = (const float*)d_in[7];
  const float* Whh1  = (const float*)d_in[8];
  const float* bih1  = (const float*)d_in[9];
  const float* bhh1  = (const float*)d_in[10];
  const float* dWih  = (const float*)d_in[11];
  const float* dWhh  = (const float*)d_in[12];
  const float* dbih  = (const float*)d_in[13];
  const float* dbhh  = (const float*)d_in[14];
  const float* fcW   = (const float*)d_in[15];
  const float* fcb   = (const float*)d_in[16];
  float* out = (float*)d_out;

  // ----- workspace carve-out (aligned to 256B) -----
  char* ws = (char*)d_ws;
  size_t off = 0;
  auto alloc = [&](size_t bytes) -> void* {
    void* p = ws + off;
    off += (bytes + 255) & ~(size_t)255;
    return p;
  };
  unsigned short* wtIh0 = (unsigned short*)alloc((size_t)Ee * G4H * 2);          // (512 x 4096)
  unsigned short* wtHh0 = (unsigned short*)alloc((size_t)Hh * G4H * 2);          // (1024 x 4096)
  unsigned short* wtIh1 = (unsigned short*)alloc((size_t)Hh * G4H * 2);
  unsigned short* wtHh1 = (unsigned short*)alloc((size_t)Hh * G4H * 2);
  unsigned short* wtDIh = (unsigned short*)alloc((size_t)(Ee + Hh) * G4H * 2);   // (1536 x 4096)
  unsigned short* wtDHh = (unsigned short*)alloc((size_t)Hh * G4H * 2);
  unsigned short* wtFc  = (unsigned short*)alloc((size_t)Hh * Vv * 2);           // (1024 x 32000)
  unsigned short* embB  = (unsigned short*)alloc((size_t)Ss * Bb * Ee * 2);      // [t][b][e]
  float*          G     = (float*)alloc((size_t)Ss * Bb * G4H * 4);              // gate pre-acts
  unsigned short* hs0   = (unsigned short*)alloc((size_t)Ss * Bb * Hh * 2);      // [t][b][h]
  unsigned short* hs1   = (unsigned short*)alloc((size_t)Ss * Bb * Hh * 2);
  unsigned short* hdec  = (unsigned short*)alloc((size_t)Bb * Hh * 2);
  unsigned short* ctxB  = (unsigned short*)alloc((size_t)Bb * Hh * 2);
  float*          c0    = (float*)alloc((size_t)Bb * Hh * 4);
  float*          c1    = (float*)alloc((size_t)Bb * Hh * 4);
  float*          sc    = (float*)alloc((size_t)Bb * Ss * 4);
  float*          aw    = (float*)alloc((size_t)Bb * Ss * 4);

  auto gemm = [&](const unsigned short* A, int lda, const unsigned short* Wt, int ldb,
                  float* C, long ldc, const float* bias, int acc,
                  int M, int N, int K) {
    dim3 grid(N / 256, M / 32);
    gemm_wmma_bf16<<<grid, 256, 0, stream>>>(A, lda, Wt, ldb, C, ldc, bias, acc, K);
  };
  auto ct = [&](const float* in, unsigned short* outp, int N, int K) {
    long tot = (long)N * K;
    convert_transpose<<<(int)((tot + 255) / 256), 256, 0, stream>>>(in, outp, N, K, tot);
  };

  // ----- one-time weight conversion + transpose to (K x N) bf16 -----
  ct(Wih0, wtIh0, G4H, Ee);
  ct(Whh0, wtHh0, G4H, Hh);
  ct(Wih1, wtIh1, G4H, Hh);
  ct(Whh1, wtHh1, G4H, Hh);
  ct(dWih, wtDIh, G4H, Ee + Hh);
  ct(dWhh, wtDHh, G4H, Hh);
  ct(fcW,  wtFc,  Vv,  Hh);

  hipMemsetAsync(c0, 0, (size_t)Bb * Hh * 4, stream);
  hipMemsetAsync(c1, 0, (size_t)Bb * Hh * 4, stream);

  // =========================== encoder ===========================
  {
    long tot = (long)Ss * Bb * Ee;
    embed_lookup<<<(int)((tot + 255) / 256), 256, 0, stream>>>(src, Ss, embed, embB, tot);
  }
  // layer 0: hoisted input projection (M = 4096), then recurrent scan
  gemm(embB, Ee, wtIh0, G4H, G, G4H, nullptr, 0, Ss * Bb, G4H, Ee);
  for (int t = 0; t < Ss; ++t) {
    float* Gt = G + (long)t * Bb * G4H;
    if (t > 0)
      gemm(hs0 + (long)(t - 1) * Bb * Hh, Hh, wtHh0, G4H, Gt, G4H, nullptr, 1, Bb, G4H, Hh);
    lstm_pointwise<<<(Bb * Hh + 255) / 256, 256, 0, stream>>>(Gt, bih0, bhh0, c0,
                                                              hs0 + (long)t * Bb * Hh);
  }
  // layer 1
  gemm(hs0, Hh, wtIh1, G4H, G, G4H, nullptr, 0, Ss * Bb, G4H, Hh);
  for (int t = 0; t < Ss; ++t) {
    float* Gt = G + (long)t * Bb * G4H;
    if (t > 0)
      gemm(hs1 + (long)(t - 1) * Bb * Hh, Hh, wtHh1, G4H, Gt, G4H, nullptr, 1, Bb, G4H, Hh);
    lstm_pointwise<<<(Bb * Hh + 255) / 256, 256, 0, stream>>>(Gt, bih1, bhh1, c1,
                                                              hs1 + (long)t * Bb * Hh);
  }

  // =========================== decoder ===========================
  {
    long tot = (long)(Tt - 1) * Bb * Ee;
    embed_lookup<<<(int)((tot + 255) / 256), 256, 0, stream>>>(tgt, Tt, embed, embB, tot);
  }
  // hoisted x_t part of dec_Wih (columns 0..E-1 => rows 0..E-1 of Wt)
  gemm(embB, Ee, wtDIh, G4H, G, G4H, nullptr, 0, (Tt - 1) * Bb, G4H, Ee);

  for (int t = 0; t < Tt - 1; ++t) {
    const unsigned short* hprev = (t == 0) ? hs1 + (long)(Ss - 1) * Bb * Hh : hdec;
    attn_scores<<<(Bb * Ss) / 8, 256, 0, stream>>>(hs1, hprev, sc);
    softmax128<<<Bb, Ss, 0, stream>>>(sc, aw);
    attn_ctx<<<(Bb * Hh) / 256, 256, 0, stream>>>(aw, hs1, ctxB);

    float* Gt = G + (long)t * Bb * G4H;
    // ctx part of dec_Wih: columns E..E+H-1 => Wt rows E..E+H-1
    gemm(ctxB, Hh, wtDIh + (size_t)Ee * G4H, G4H, Gt, G4H, nullptr, 1, Bb, G4H, Hh);
    gemm(hprev, Hh, wtDHh, G4H, Gt, G4H, nullptr, 1, Bb, G4H, Hh);
    lstm_pointwise<<<(Bb * Hh + 255) / 256, 256, 0, stream>>>(Gt, dbih, dbhh, c1, hdec);

    // logits: out[b][t][:] ; row stride (Tt-1)*V, base offset t*V
    gemm(hdec, Hh, wtFc, Vv, out + (long)t * Vv, (long)(Tt - 1) * Vv, fcb, 0, Bb, Vv, Hh);
  }
}
